// BipartiteGNNLayer_59365037965999
// MI455X (gfx1250) — matrix operand order repaired and checked by reading, not compiled, not run on hardware
//
#include <hip/hip_runtime.h>
#include <hip/hip_bf16.h>

// ---------------------------------------------------------------------------
// Bipartite GNN layer (GINEConv x2 + LayerNorm x2) for MI455X / gfx1250.
// H=256, NV=NC=50000, E=300000, ED=1.
// GEMMs run on v_wmma_f32_16x16x32_bf16 (fp32 accumulate); edge phase uses
// global_atomic_add_f32; LayerNorm uses wave32 shuffles and applies residual.
// ---------------------------------------------------------------------------

#define H 256
#define MB 128            // rows per workgroup in the fused-MLP kernel
#define LDP 264           // padded LDS stride (bf16 elems): kills 512B-stride bank conflicts
#define LN_EPS 1e-5f

typedef __bf16 v16bf __attribute__((ext_vector_type(16)));
typedef float  v8f   __attribute__((ext_vector_type(8)));

static __device__ __forceinline__ __bf16 f2bf(float f) {
    unsigned u = __builtin_bit_cast(unsigned, f);
    unsigned r = u + 0x7FFFu + ((u >> 16) & 1u);   // round-to-nearest-even
    unsigned short h = (unsigned short)(r >> 16);
    return __builtin_bit_cast(__bf16, h);
}

// --- zero fill ---------------------------------------------------------------
__global__ void zero_kernel(float* __restrict__ p, int n) {
    int i = blockIdx.x * blockDim.x + threadIdx.x;
    if (i < n) p[i] = 0.0f;
}

// --- weight convert: out[m][n*256+k] = bf16(W_m[k*256+n])  (transposed) ------
__global__ void wcvt4_kernel(const float* __restrict__ W0, const float* __restrict__ W1,
                             const float* __restrict__ W2, const float* __restrict__ W3,
                             __bf16* __restrict__ T) {
    const float* W = (blockIdx.y == 0) ? W0 : (blockIdx.y == 1) ? W1
                   : (blockIdx.y == 2) ? W2 : W3;
    __bf16* out = T + (size_t)blockIdx.y * (H * H);
    int n = blockIdx.x;          // output row (= W column)
    int k = threadIdx.x;         // output col (= W row)
    out[(size_t)n * H + k] = f2bf(W[(size_t)k * H + n]);
}

// --- edge scatter: agg[dst] += relu(x_src[src] + a_e*We + be) ----------------
// one wave per edge; lane handles 8 columns (two float4 loads)
__global__ void scatter_kernel(const float* __restrict__ xsrc,
                               const int* __restrict__ eidx,    // [2,E]
                               const float* __restrict__ eattr, // [E]
                               const float* __restrict__ We,    // [256]
                               const float* __restrict__ be,    // [256]
                               float* __restrict__ agg, int E) {
    int tid  = threadIdx.x;
    int e    = blockIdx.x * 8 + (tid >> 5);
    int lane = tid & 31;
    if (e >= E) return;
    int src = eidx[e];
    int dst = eidx[E + e];
    float a = eattr[e];
    int c0  = lane * 8;
    const float4 x0 = *(const float4*)(xsrc + (size_t)src * H + c0);
    const float4 x1 = *(const float4*)(xsrc + (size_t)src * H + c0 + 4);
    const float4 w0 = *(const float4*)(We + c0);
    const float4 w1 = *(const float4*)(We + c0 + 4);
    const float4 b0 = *(const float4*)(be + c0);
    const float4 b1 = *(const float4*)(be + c0 + 4);
    float m[8];
    m[0] = x0.x + a * w0.x + b0.x;  m[1] = x0.y + a * w0.y + b0.y;
    m[2] = x0.z + a * w0.z + b0.z;  m[3] = x0.w + a * w0.w + b0.w;
    m[4] = x1.x + a * w1.x + b1.x;  m[5] = x1.y + a * w1.y + b1.y;
    m[6] = x1.z + a * w1.z + b1.z;  m[7] = x1.w + a * w1.w + b1.w;
    float* dptr = agg + (size_t)dst * H + c0;
#pragma unroll
    for (int i = 0; i < 8; i++) {
        float v = m[i] > 0.0f ? m[i] : 0.0f;
        atomicAdd(dptr + i, v);
    }
}

// --- fused 2-layer MLP with WMMA ---------------------------------------------
// out[row] = relu((xin[row]+agg[row]) @ Wa + ba) @ Wb + bb    (no residual;
// the LayerNorm kernel adds xin).  WaT/WbT are transposed bf16 ([n][k]).
__global__ void mlp_kernel(const float* __restrict__ xin,
                           const float* __restrict__ agg,
                           const __bf16* __restrict__ WaT,
                           const float* __restrict__ ba,
                           const __bf16* __restrict__ WbT,
                           const float* __restrict__ bb,
                           float* __restrict__ out, int N) {
    extern __shared__ __bf16 smem[];
    __bf16* sW = smem;                 // 256 x LDP  (weights, W^T)
    __bf16* sA = smem + H * LDP;       // MB  x LDP  (activations h, then t)

    const int tid  = threadIdx.x;
    const int base = blockIdx.x * MB;

    // pull stage-2 weights toward cache early (global_prefetch_b8)
    __builtin_prefetch(WbT, 0, 1);
    __builtin_prefetch(WbT + 32 * H, 0, 1);

    // fill activation tile: bf16(xin + agg), zeros past N
    for (int idx = tid; idx < MB * (H / 4); idx += blockDim.x) {
        int row = idx >> 6;            // H/4 == 64 quads per row
        int q   = idx & 63;
        int grow = base + row;
        float4 h = make_float4(0.f, 0.f, 0.f, 0.f);
        if (grow < N) {
            const float4 x4 = *(const float4*)(xin + (size_t)grow * H + q * 4);
            const float4 a4 = *(const float4*)(agg + (size_t)grow * H + q * 4);
            h.x = x4.x + a4.x; h.y = x4.y + a4.y;
            h.z = x4.z + a4.z; h.w = x4.w + a4.w;
        }
        __bf16* d = sA + row * LDP + q * 4;
        d[0] = f2bf(h.x); d[1] = f2bf(h.y); d[2] = f2bf(h.z); d[3] = f2bf(h.w);
    }
    // load Wa^T into LDS
    for (int idx = tid; idx < H * (H / 8); idx += blockDim.x) {
        int n = idx >> 5;              // H/8 == 32 octets per row
        int o = (idx & 31) * 8;
        const __bf16* s = WaT + (size_t)n * H + o;
        __bf16* d = sW + n * LDP + o;
#pragma unroll
        for (int t = 0; t < 8; t++) d[t] = s[t];
    }
    __syncthreads();

    const int wave = tid >> 5;
    const int lane = tid & 31;
    const int half = lane >> 4;
    const int l16  = lane & 15;
    const int lr0  = wave * 16;        // this wave's local row base

    // hoist per-lane biases out of the tile loops (one wait, off critical path)
    float bias_a[16], bias_b[16];
#pragma unroll
    for (int nt = 0; nt < 16; nt++) bias_a[nt] = ba[nt * 16 + l16];
#pragma unroll
    for (int nt = 0; nt < 16; nt++) bias_b[nt] = bb[nt * 16 + l16];

    // ---------------- stage 1: t = relu(h @ Wa + ba) ----------------
    v16bf afr[8];
#pragma unroll
    for (int kk = 0; kk < 8; kk++) {
        const __bf16* ap = sA + (lr0 + l16) * LDP + kk * 32 + half * 16;
#pragma unroll
        for (int j = 0; j < 16; j++) {
            int k = j;                       // A layout: K = half*16+j
            afr[kk][j] = ap[k];
        }
    }
    for (int nt = 0; nt < 16; nt++) {
        const __bf16* bbase = sW + (nt * 16 + l16) * LDP + half * 16;
        v16bf bcur;
#pragma unroll
        for (int j = 0; j < 16; j++) bcur[j] = bbase[j];   // B layout: K = half*16+j
        v8f acc;
#pragma unroll
        for (int i = 0; i < 8; i++) acc[i] = bias_a[nt];
#pragma unroll
        for (int kk = 0; kk < 8; kk++) {
            v16bf bnext;                // double-buffer: load B[kk+1] before WMMA[kk]
            if (kk < 7) {
                const __bf16* bp = bbase + (kk + 1) * 32;
#pragma unroll
                for (int j = 0; j < 16; j++) bnext[j] = bp[j];
            }
            acc = __builtin_amdgcn_wmma_f32_16x16x32_bf16(
                false, afr[kk], false, bcur, (short)0, acc, false, false);
            if (kk < 7) bcur = bnext;
        }
        // relu -> bf16 back into sA (own rows only; A-frags already in regs)
#pragma unroll
        for (int i = 0; i < 8; i++) {
            float v = acc[i];
            v = v > 0.0f ? v : 0.0f;
            sA[(lr0 + half * 8 + i) * LDP + nt * 16 + l16] = f2bf(v);
        }
    }
    __syncthreads();
    // swap in Wb^T
    for (int idx = tid; idx < H * (H / 8); idx += blockDim.x) {
        int n = idx >> 5;
        int o = (idx & 31) * 8;
        const __bf16* s = WbT + (size_t)n * H + o;
        __bf16* d = sW + n * LDP + o;
#pragma unroll
        for (int t = 0; t < 8; t++) d[t] = s[t];
    }
    __syncthreads();

    // ---------------- stage 2: out = t @ Wb + bb ----------------
#pragma unroll
    for (int kk = 0; kk < 8; kk++) {
        const __bf16* ap = sA + (lr0 + l16) * LDP + kk * 32 + half * 16;
#pragma unroll
        for (int j = 0; j < 16; j++) {
            int k = j;
            afr[kk][j] = ap[k];
        }
    }
    for (int nt = 0; nt < 16; nt++) {
        const __bf16* bbase = sW + (nt * 16 + l16) * LDP + half * 16;
        v16bf bcur;
#pragma unroll
        for (int j = 0; j < 16; j++) bcur[j] = bbase[j];
        v8f acc;
#pragma unroll
        for (int i = 0; i < 8; i++) acc[i] = bias_b[nt];
#pragma unroll
        for (int kk = 0; kk < 8; kk++) {
            v16bf bnext;
            if (kk < 7) {
                const __bf16* bp = bbase + (kk + 1) * 32;
#pragma unroll
                for (int j = 0; j < 16; j++) bnext[j] = bp[j];
            }
            acc = __builtin_amdgcn_wmma_f32_16x16x32_bf16(
                false, afr[kk], false, bcur, (short)0, acc, false, false);
            if (kk < 7) bcur = bnext;
        }
        int col = nt * 16 + l16;
#pragma unroll
        for (int i = 0; i < 8; i++) {
            int row = base + lr0 + half * 8 + i;
            if (row < N) out[(size_t)row * H + col] = acc[i];
        }
    }
}

// --- in-place LayerNorm with residual: y = LN(x + y) -------------------------
// one wave per row; coalesced float4 traffic
__global__ void ln_kernel(float* __restrict__ y, const float* __restrict__ x,
                          const float* __restrict__ g, const float* __restrict__ b,
                          int N) {
    int tid  = threadIdx.x;
    int row  = blockIdx.x * 8 + (tid >> 5);
    int lane = tid & 31;
    if (row >= N) return;
    float*       p  = y + (size_t)row * H + lane * 8;
    const float* xp = x + (size_t)row * H + lane * 8;
    float4 v0 = *(float4*)(p);
    float4 v1 = *(float4*)(p + 4);
    const float4 r0 = *(const float4*)(xp);
    const float4 r1 = *(const float4*)(xp + 4);
    v0.x += r0.x; v0.y += r0.y; v0.z += r0.z; v0.w += r0.w;
    v1.x += r1.x; v1.y += r1.y; v1.z += r1.z; v1.w += r1.w;
    float s  = v0.x + v0.y + v0.z + v0.w + v1.x + v1.y + v1.z + v1.w;
    float sq = v0.x * v0.x + v0.y * v0.y + v0.z * v0.z + v0.w * v0.w
             + v1.x * v1.x + v1.y * v1.y + v1.z * v1.z + v1.w * v1.w;
#pragma unroll
    for (int m = 16; m >= 1; m >>= 1) {
        s  += __shfl_xor(s,  m, 32);
        sq += __shfl_xor(sq, m, 32);
    }
    float mu  = s * (1.0f / H);
    float var = sq * (1.0f / H) - mu * mu;
    float rs  = rsqrtf(var + LN_EPS);
    int c = lane * 8;
    const float4 g0 = *(const float4*)(g + c), g1 = *(const float4*)(g + c + 4);
    const float4 b0 = *(const float4*)(b + c), b1 = *(const float4*)(b + c + 4);
    float4 o0, o1;
    o0.x = (v0.x - mu) * rs * g0.x + b0.x;  o0.y = (v0.y - mu) * rs * g0.y + b0.y;
    o0.z = (v0.z - mu) * rs * g0.z + b0.z;  o0.w = (v0.w - mu) * rs * g0.w + b0.w;
    o1.x = (v1.x - mu) * rs * g1.x + b1.x;  o1.y = (v1.y - mu) * rs * g1.y + b1.y;
    o1.z = (v1.z - mu) * rs * g1.z + b1.z;  o1.w = (v1.w - mu) * rs * g1.w + b1.w;
    *(float4*)(p)     = o0;
    *(float4*)(p + 4) = o1;
}

// ---------------------------------------------------------------------------
extern "C" void kernel_launch(void* const* d_in, const int* in_sizes, int n_in,
                              void* d_out, int out_size, void* d_ws, size_t ws_size,
                              hipStream_t stream) {
    const float* x_var    = (const float*)d_in[0];
    const float* x_constr = (const float*)d_in[1];
    const int*   ei_v2c   = (const int*)d_in[2];
    const int*   ei_c2v   = (const int*)d_in[3];
    const float* eattr    = (const float*)d_in[4];
    const float* We1 = (const float*)d_in[5],  *be1 = (const float*)d_in[6];
    const float* W1a = (const float*)d_in[7],  *b1a = (const float*)d_in[8];
    const float* W1b = (const float*)d_in[9],  *b1b = (const float*)d_in[10];
    const float* We2 = (const float*)d_in[11], *be2 = (const float*)d_in[12];
    const float* W2a = (const float*)d_in[13], *b2a = (const float*)d_in[14];
    const float* W2b = (const float*)d_in[15], *b2b = (const float*)d_in[16];
    const float* g_c = (const float*)d_in[17], *bt_c = (const float*)d_in[18];
    const float* g_v = (const float*)d_in[19], *bt_v = (const float*)d_in[20];

    const int NV = in_sizes[0] / H;
    const int NC = in_sizes[1] / H;
    const int E  = in_sizes[2] / 2;
    const int NMAX = NV > NC ? NV : NC;

    float* out_var    = (float*)d_out;                      // tuple order: x_var first
    float* out_constr = (float*)d_out + (size_t)NV * H;

    char*   wsb  = (char*)d_ws;
    float*  agg  = (float*)wsb;                              // NMAX*H f32
    size_t  aggB = (size_t)NMAX * H * sizeof(float);
    __bf16* wT   = (__bf16*)(wsb + aggB);                    // 4 * H*H bf16 (transposed)
    __bf16* wT1a = wT;
    __bf16* wT1b = wT + 1 * (size_t)H * H;
    __bf16* wT2a = wT + 2 * (size_t)H * H;
    __bf16* wT2b = wT + 3 * (size_t)H * H;

    const size_t SMEM = (size_t)(H * LDP + MB * LDP) * sizeof(__bf16); // ~198 KB

    // 0) weights -> transposed bf16
    wcvt4_kernel<<<dim3(H, 4), H, 0, stream>>>(W1a, W1b, W2a, W2b, wT);

    // 1) GINE var->constr
    zero_kernel<<<(NC * H + 255) / 256, 256, 0, stream>>>(agg, NC * H);
    scatter_kernel<<<(E + 7) / 8, 256, 0, stream>>>(x_var, ei_v2c, eattr, We1, be1, agg, E);
    mlp_kernel<<<(NC + MB - 1) / MB, 256, SMEM, stream>>>(x_constr, agg, wT1a, b1a,
                                                          wT1b, b1b, out_constr, NC);
    ln_kernel<<<(NC + 7) / 8, 256, 0, stream>>>(out_constr, x_constr, g_c, bt_c, NC);

    // 2) GINE constr->var (uses LayerNorm'd constr features)
    zero_kernel<<<(NV * H + 255) / 256, 256, 0, stream>>>(agg, NV * H);
    scatter_kernel<<<(E + 7) / 8, 256, 0, stream>>>(out_constr, ei_c2v, eattr, We2, be2, agg, E);
    mlp_kernel<<<(NV + MB - 1) / MB, 256, SMEM, stream>>>(x_var, agg, wT2a, b2a,
                                                          wT2b, b2b, out_var, NV);
    ln_kernel<<<(NV + 7) / 8, 256, 0, stream>>>(out_var, x_var, g_v, bt_v, NV);
}
